// MultiHeadAttention_1967095021855
// MI455X (gfx1250) — matrix operand order repaired and checked by reading, not compiled, not run on hardware
//
#include <hip/hip_runtime.h>
#include <cstdint>

// Problem constants (match reference)
#define B_    2
#define S_    2048
#define DIN_  1024
#define DOUT_ 1024
#define H_    16
#define DH_   64
#define M_    (B_ * S_)          // 4096 rows of x
#define BH_   (B_ * H_)          // 32

typedef __attribute__((ext_vector_type(16))) __bf16 bf16x16;
typedef __attribute__((ext_vector_type(8)))  float  f32x8;

union Frag {
  bf16x16  v;
  uint32_t u[8];
};

__device__ __forceinline__ uint16_t f2bf(float f) {
  union { float f; uint32_t u; } c; c.f = f;
  uint32_t u = c.u;
  u += 0x7FFFu + ((u >> 16) & 1u);   // round to nearest even
  return (uint16_t)(u >> 16);
}
__device__ __forceinline__ uint32_t pack_bf2(float lo, float hi) {
  return (uint32_t)f2bf(lo) | ((uint32_t)f2bf(hi) << 16);
}

__device__ __forceinline__ f32x8 wmma_bf16(const Frag& a, const Frag& b, f32x8 c) {
  // D = A(16x32 bf16) * B(32x16 bf16) + C(16x16 f32)
  return __builtin_amdgcn_wmma_f32_16x16x32_bf16(
      /*neg_a=*/false, a.v, /*neg_b=*/false, b.v,
      /*c_mod=*/(short)0, c, /*reuse_a=*/false, /*reuse_b=*/false);
}

// A fragment: 16(M) x 32(K) bf16 from row-major storage (row stride ld elements).
// lane L -> row m0+(L&15); VGPR v -> packed K pair at (v/4)*16 + (v%4)*2 + 8*(L>=16)
__device__ __forceinline__ Frag load_a_frag(const uint16_t* base, int ld, int m0, int k0, int lane) {
  Frag f;
  const uint16_t* r = base + (size_t)(m0 + (lane & 15)) * ld + k0 + ((lane >> 4) << 3);
#pragma unroll
  for (int v = 0; v < 8; ++v) {
    int k = ((v >> 2) << 4) + ((v & 3) << 1);
    f.u[v] = *(const uint32_t*)(r + k);
  }
  return f;
}

// B fragment: 32(K) x 16(N) bf16 where B[k][n] is stored transposed as T[n][k]
// (k contiguous). lane L -> col n0+(L&15); VGPR v -> packed K pair at (v/4)*16 + (v%4)*2 + 8*(L>=16)
__device__ __forceinline__ Frag load_b_frag(const uint16_t* baseT, int ld, int n0, int k0, int lane) {
  Frag f;
  const uint16_t* r = baseT + (size_t)(n0 + (lane & 15)) * ld + k0 + ((lane >> 4) << 3);
#pragma unroll
  for (int v = 0; v < 8; ++v)
    f.u[v] = *(const uint32_t*)(r + ((v >> 2) << 4) + ((v & 3) << 1));
  return f;
}

// ---------------------------------------------------------------------------
// Precision conversion / weight transpose
// ---------------------------------------------------------------------------
__global__ void cvt_f32_bf16(const float* __restrict__ in, uint16_t* __restrict__ out, int n) {
  int i = blockIdx.x * blockDim.x + threadIdx.x;
  int stride = gridDim.x * blockDim.x;
  for (; i < n; i += stride) out[i] = f2bf(in[i]);
}

// in: [K=1024][N=1024] f32 row-major  ->  out: [N][K] bf16 (k contiguous)
__global__ void transpose_w_bf16(const float* __restrict__ in, uint16_t* __restrict__ out) {
  int i = blockIdx.x * blockDim.x + threadIdx.x;   // 0..1048575
  int n = i & 1023;
  int k = i >> 10;
  out[(size_t)n * 1024 + k] = f2bf(in[(size_t)k * 1024 + n]);
}

// ---------------------------------------------------------------------------
// Shared GEMM core: 32(M) x 64(N) tile per wave over K=1024, explicit
// ping-pong double buffering (two k-steps per loop iteration, no reg copies).
// Result left in acc[8] (acc[0..3] = rows m0..m0+15, acc[4..7] = rows +16).
// ---------------------------------------------------------------------------
__device__ __forceinline__ void gemm_tile_1024(const uint16_t* __restrict__ A,
                                               const uint16_t* __restrict__ Wt,
                                               int m0, int n0, int lane, f32x8 acc[8]) {
#pragma unroll
  for (int j = 0; j < 8; ++j) acc[j] = (f32x8)0.0f;

  // Set X <- k = 0
  Frag xa0 = load_a_frag(A, DIN_, m0,      0, lane);
  Frag xa1 = load_a_frag(A, DIN_, m0 + 16, 0, lane);
  Frag xb0 = load_b_frag(Wt, DIN_, n0,      0, lane);
  Frag xb1 = load_b_frag(Wt, DIN_, n0 + 16, 0, lane);
  Frag xb2 = load_b_frag(Wt, DIN_, n0 + 32, 0, lane);
  Frag xb3 = load_b_frag(Wt, DIN_, n0 + 48, 0, lane);

#pragma unroll 1
  for (int k0 = 0; k0 < DIN_ - 64; k0 += 64) {
    // Set Y <- k0+32 (issued before consuming X)
    Frag ya0 = load_a_frag(A, DIN_, m0,      k0 + 32, lane);
    Frag ya1 = load_a_frag(A, DIN_, m0 + 16, k0 + 32, lane);
    Frag yb0 = load_b_frag(Wt, DIN_, n0,      k0 + 32, lane);
    Frag yb1 = load_b_frag(Wt, DIN_, n0 + 16, k0 + 32, lane);
    Frag yb2 = load_b_frag(Wt, DIN_, n0 + 32, k0 + 32, lane);
    Frag yb3 = load_b_frag(Wt, DIN_, n0 + 48, k0 + 32, lane);
    __builtin_prefetch(A + (size_t)(m0 + (lane & 15)) * DIN_ + k0 + 64, 0, 3);

    acc[0] = wmma_bf16(xa0, xb0, acc[0]);
    acc[4] = wmma_bf16(xa1, xb0, acc[4]);
    acc[1] = wmma_bf16(xa0, xb1, acc[1]);
    acc[5] = wmma_bf16(xa1, xb1, acc[5]);
    acc[2] = wmma_bf16(xa0, xb2, acc[2]);
    acc[6] = wmma_bf16(xa1, xb2, acc[6]);
    acc[3] = wmma_bf16(xa0, xb3, acc[3]);
    acc[7] = wmma_bf16(xa1, xb3, acc[7]);

    // Set X <- k0+64 (issued before consuming Y)
    xa0 = load_a_frag(A, DIN_, m0,      k0 + 64, lane);
    xa1 = load_a_frag(A, DIN_, m0 + 16, k0 + 64, lane);
    xb0 = load_b_frag(Wt, DIN_, n0,      k0 + 64, lane);
    xb1 = load_b_frag(Wt, DIN_, n0 + 16, k0 + 64, lane);
    xb2 = load_b_frag(Wt, DIN_, n0 + 32, k0 + 64, lane);
    xb3 = load_b_frag(Wt, DIN_, n0 + 48, k0 + 64, lane);

    acc[0] = wmma_bf16(ya0, yb0, acc[0]);
    acc[4] = wmma_bf16(ya1, yb0, acc[4]);
    acc[1] = wmma_bf16(ya0, yb1, acc[1]);
    acc[5] = wmma_bf16(ya1, yb1, acc[5]);
    acc[2] = wmma_bf16(ya0, yb2, acc[2]);
    acc[6] = wmma_bf16(ya1, yb2, acc[6]);
    acc[3] = wmma_bf16(ya0, yb3, acc[3]);
    acc[7] = wmma_bf16(ya1, yb3, acc[7]);
  }

  // Epilogue: X holds k = 960; load Y for k = 992, then consume both.
  {
    Frag ya0 = load_a_frag(A, DIN_, m0,      DIN_ - 32, lane);
    Frag ya1 = load_a_frag(A, DIN_, m0 + 16, DIN_ - 32, lane);
    Frag yb0 = load_b_frag(Wt, DIN_, n0,      DIN_ - 32, lane);
    Frag yb1 = load_b_frag(Wt, DIN_, n0 + 16, DIN_ - 32, lane);
    Frag yb2 = load_b_frag(Wt, DIN_, n0 + 32, DIN_ - 32, lane);
    Frag yb3 = load_b_frag(Wt, DIN_, n0 + 48, DIN_ - 32, lane);

    acc[0] = wmma_bf16(xa0, xb0, acc[0]);
    acc[4] = wmma_bf16(xa1, xb0, acc[4]);
    acc[1] = wmma_bf16(xa0, xb1, acc[1]);
    acc[5] = wmma_bf16(xa1, xb1, acc[5]);
    acc[2] = wmma_bf16(xa0, xb2, acc[2]);
    acc[6] = wmma_bf16(xa1, xb2, acc[6]);
    acc[3] = wmma_bf16(xa0, xb3, acc[3]);
    acc[7] = wmma_bf16(xa1, xb3, acc[7]);

    acc[0] = wmma_bf16(ya0, yb0, acc[0]);
    acc[4] = wmma_bf16(ya1, yb0, acc[4]);
    acc[1] = wmma_bf16(ya0, yb1, acc[1]);
    acc[5] = wmma_bf16(ya1, yb1, acc[5]);
    acc[2] = wmma_bf16(ya0, yb2, acc[2]);
    acc[6] = wmma_bf16(ya1, yb2, acc[6]);
    acc[3] = wmma_bf16(ya0, yb3, acc[3]);
    acc[7] = wmma_bf16(ya1, yb3, acc[7]);
  }
}

// ---------------------------------------------------------------------------
// QKV projection GEMM: C[M_,1024] = xb[M_,1024] @ W  (W given as Wt[n][k] bf16)
// mode 0: store bf16 to [BH][S][DH] (Q, K).  mode 1: store bf16 to [BH][DH][S] (V^T).
// ---------------------------------------------------------------------------
__global__ void __launch_bounds__(256, 1)
gemm_qkv(const uint16_t* __restrict__ A, const uint16_t* __restrict__ Wt,
         uint16_t* __restrict__ out, int mode) {
  int lane = threadIdx.x & 31;
  int wid  = blockIdx.x * (blockDim.x >> 5) + (threadIdx.x >> 5);
  int mt = wid >> 4;            // 0..127
  int nt = wid & 15;            // 0..15
  int m0 = mt * 32, n0 = nt * 64;

  f32x8 acc[8];
  gemm_tile_1024(A, Wt, m0, n0, lane, acc);

  int roff = (lane >> 4) << 3;      // 0 or 8
#pragma unroll
  for (int t = 0; t < 2; ++t) {
#pragma unroll
    for (int j = 0; j < 4; ++j) {
#pragma unroll
      for (int v = 0; v < 8; ++v) {
        int m = m0 + 16 * t + v + roff;
        int n = n0 + 16 * j + (lane & 15);
        int bb = m >> 11, s = m & (S_ - 1);
        int h  = n >> 6,  d = n & (DH_ - 1);
        size_t idx = (mode == 0)
            ? ((size_t)(bb * H_ + h) * S_ + s) * DH_ + d          // [BH][S][DH]
            : ((size_t)(bb * H_ + h) * DH_ + d) * S_ + s;         // [BH][DH][S]
        out[idx] = f2bf(acc[4 * t + j][v]);
      }
    }
  }
}

// ---------------------------------------------------------------------------
// Flash attention (causal), one wave per (bh, 16-query tile).
// Scores computed TRANSPOSED: T = K_tile(16k x 64d) x Q^T(64d x 16q) so that the
// P -> A-fragment conversion for P·V is lane-local (no cross-lane shuffle).
// ---------------------------------------------------------------------------
__global__ void __launch_bounds__(128, 1)
flash_attn(const uint16_t* __restrict__ Q, const uint16_t* __restrict__ K,
           const uint16_t* __restrict__ Vt, uint16_t* __restrict__ co) {
  int lane = threadIdx.x & 31;
  int wid  = blockIdx.x * (blockDim.x >> 5) + (threadIdx.x >> 5);
  int bh = wid >> 7;                 // 0..31
  int qt = wid & 127;                // 0..127
  int q0 = qt * 16;

  const uint16_t* Qh = Q  + (size_t)bh * S_ * DH_;
  const uint16_t* Kh = K  + (size_t)bh * S_ * DH_;
  const uint16_t* Vh = Vt + (size_t)bh * DH_ * S_;

  f32x8 acc[4];
#pragma unroll
  for (int j = 0; j < 4; ++j) acc[j] = (f32x8)0.0f;
  float mrun = -INFINITY, lrun = 0.0f;

  const float scale = 0.022097086912079608f;   // 1/sqrt(S_) — faithful to reference
  int qg   = q0 + (lane & 15);                 // this lane's query (column of S^T)
  int roff = (lane >> 4) << 3;                 // 0 or 8 (row offset in C frags)

  // Q^T B-fragments are loop-invariant: hoist out of the key loop.
  Frag bQ0 = load_b_frag(Qh, DH_, q0, 0,  lane);
  Frag bQ1 = load_b_frag(Qh, DH_, q0, 32, lane);

  int nkc = (qt >> 1) + 1;                     // causal: 32-key chunks needed
#pragma unroll 1
  for (int kc = 0; kc < nkc; ++kc) {
    int kb = kc * 32;

    // ---- scores^T: two 16key x 16query tiles, contracting over DH=64 ----
    Frag aK0 = load_a_frag(Kh, DH_, kb,      0,  lane);
    Frag aK1 = load_a_frag(Kh, DH_, kb + 16, 0,  lane);
    Frag aK2 = load_a_frag(Kh, DH_, kb,      32, lane);
    Frag aK3 = load_a_frag(Kh, DH_, kb + 16, 32, lane);
    f32x8 t0 = (f32x8)0.0f, t1 = (f32x8)0.0f;
    t0 = wmma_bf16(aK0, bQ0, t0);
    t1 = wmma_bf16(aK1, bQ0, t1);
    t0 = wmma_bf16(aK2, bQ1, t0);
    t1 = wmma_bf16(aK3, bQ1, t1);

    // Issue V loads now so they overlap the softmax VALU work below.
    Frag bV0 = load_b_frag(Vh, S_, 0,  kb, lane);
    Frag bV1 = load_b_frag(Vh, S_, 16, kb, lane);
    Frag bV2 = load_b_frag(Vh, S_, 32, kb, lane);
    Frag bV3 = load_b_frag(Vh, S_, 48, kb, lane);

    // ---- scale + causal mask + online softmax ----
    float p0[8], p1[8];
    float lm = -INFINITY;
#pragma unroll
    for (int v = 0; v < 8; ++v) {
      int k0g = kb + v + roff;            // key row of t0[v]
      int k1g = kb + 16 + v + roff;       // key row of t1[v]
      float s0 = (k0g <= qg) ? t0[v] * scale : -INFINITY;
      float s1 = (k1g <= qg) ? t1[v] * scale : -INFINITY;
      p0[v] = s0; p1[v] = s1;
      lm = fmaxf(lm, fmaxf(s0, s1));
    }
    lm = fmaxf(lm, __shfl_xor(lm, 16, 32));       // combine lane pair (keys split)
    float mnew = fmaxf(mrun, lm);                 // finite: key kb is always valid
    float corr = __expf(mrun - mnew);
    float lsum = 0.0f;
#pragma unroll
    for (int v = 0; v < 8; ++v) {
      p0[v] = __expf(p0[v] - mnew);
      p1[v] = __expf(p1[v] - mnew);
      lsum += p0[v] + p1[v];
    }
    lsum += __shfl_xor(lsum, 16, 32);
    lrun = lrun * corr + lsum;
    mrun = mnew;

    // ---- rescale running context (rows = queries -> gather corr per row) ----
    float rc[8];
#pragma unroll
    for (int v = 0; v < 8; ++v) rc[v] = __shfl(corr, v + roff, 32);
#pragma unroll
    for (int j = 0; j < 4; ++j)
#pragma unroll
      for (int v = 0; v < 8; ++v) acc[j][v] *= rc[v];

    // ---- P (16q x 32k) A-fragment: lane-local pack from S^T C-fragments ----
    Frag aP;
#pragma unroll
    for (int v = 0; v < 4; ++v) {
      aP.u[v]     = pack_bf2(p0[2 * v], p0[2 * v + 1]);
      aP.u[4 + v] = pack_bf2(p1[2 * v], p1[2 * v + 1]);
    }

    // ---- ctx += P x V ----
    acc[0] = wmma_bf16(aP, bV0, acc[0]);
    acc[1] = wmma_bf16(aP, bV1, acc[1]);
    acc[2] = wmma_bf16(aP, bV2, acc[2]);
    acc[3] = wmma_bf16(aP, bV3, acc[3]);
  }

  // ---- finalize: divide by row sums, store ctx bf16 [B][S][H*DH] ----
  float rl[8];
#pragma unroll
  for (int v = 0; v < 8; ++v) rl[v] = 1.0f / __shfl(lrun, v + roff, 32);

  int b = bh >> 4, h = bh & (H_ - 1);
#pragma unroll
  for (int j = 0; j < 4; ++j) {
#pragma unroll
    for (int v = 0; v < 8; ++v) {
      int s   = q0 + v + roff;
      int col = h * DH_ + 16 * j + (lane & 15);
      co[((size_t)(b * S_ + s)) * DOUT_ + col] = f2bf(acc[j][v] * rl[v]);
    }
  }
}

// ---------------------------------------------------------------------------
// Output projection: out[M_,1024] f32 = ctx[M_,1024]bf16 @ Wo (Wot[n][k]) + bo
// ---------------------------------------------------------------------------
__global__ void __launch_bounds__(256, 1)
gemm_out(const uint16_t* __restrict__ A, const uint16_t* __restrict__ Wt,
         const float* __restrict__ bo, float* __restrict__ out) {
  int lane = threadIdx.x & 31;
  int wid  = blockIdx.x * (blockDim.x >> 5) + (threadIdx.x >> 5);
  int mt = wid >> 4;
  int nt = wid & 15;
  int m0 = mt * 32, n0 = nt * 64;

  f32x8 acc[8];
  gemm_tile_1024(A, Wt, m0, n0, lane, acc);

  int roff = (lane >> 4) << 3;
#pragma unroll
  for (int t = 0; t < 2; ++t) {
#pragma unroll
    for (int j = 0; j < 4; ++j) {
#pragma unroll
      for (int v = 0; v < 8; ++v) {
        int m = m0 + 16 * t + v + roff;
        int n = n0 + 16 * j + (lane & 15);
        out[(size_t)m * DOUT_ + n] = acc[4 * t + j][v] + bo[n];
      }
    }
  }
}

// ---------------------------------------------------------------------------
// Launch
// ---------------------------------------------------------------------------
extern "C" void kernel_launch(void* const* d_in, const int* in_sizes, int n_in,
                              void* d_out, int out_size, void* d_ws, size_t ws_size,
                              hipStream_t stream) {
  (void)in_sizes; (void)n_in; (void)out_size; (void)ws_size;
  const float* x  = (const float*)d_in[0];
  const float* Wq = (const float*)d_in[1];
  const float* Wk = (const float*)d_in[2];
  const float* Wv = (const float*)d_in[3];
  const float* Wo = (const float*)d_in[4];
  const float* bo = (const float*)d_in[5];
  float* out = (float*)d_out;

  // Workspace layout (bf16 elements), 48 MiB total
  uint16_t* w    = (uint16_t*)d_ws;
  uint16_t* xb   = w;                                   // M_*DIN
  uint16_t* Wqt  = xb  + (size_t)M_ * DIN_;             // 1024*1024
  uint16_t* Wkt  = Wqt + (size_t)DIN_ * DOUT_;
  uint16_t* Wvt  = Wkt + (size_t)DIN_ * DOUT_;
  uint16_t* Wot  = Wvt + (size_t)DIN_ * DOUT_;
  uint16_t* Qb   = Wot + (size_t)DOUT_ * DOUT_;         // [BH][S][DH]
  uint16_t* Kb   = Qb  + (size_t)BH_ * S_ * DH_;
  uint16_t* Vtb  = Kb  + (size_t)BH_ * S_ * DH_;        // [BH][DH][S]
  uint16_t* ctxb = Vtb + (size_t)BH_ * DH_ * S_;        // [M_][DOUT]

  // 1) precision conversion / weight transposes
  cvt_f32_bf16<<<4096, 256, 0, stream>>>(x, xb, M_ * DIN_);
  transpose_w_bf16<<<4096, 256, 0, stream>>>(Wq, Wqt);
  transpose_w_bf16<<<4096, 256, 0, stream>>>(Wk, Wkt);
  transpose_w_bf16<<<4096, 256, 0, stream>>>(Wv, Wvt);
  transpose_w_bf16<<<4096, 256, 0, stream>>>(Wo, Wot);

  // 2) QKV projections: 2048 waves (32x64 tiles), 8 waves/block
  gemm_qkv<<<256, 256, 0, stream>>>(xb, Wqt, Qb, 0);
  gemm_qkv<<<256, 256, 0, stream>>>(xb, Wkt, Kb, 0);
  gemm_qkv<<<256, 256, 0, stream>>>(xb, Wvt, Vtb, 1);

  // 3) causal flash attention: 32 heads * 128 query-tiles = 4096 waves
  flash_attn<<<1024, 128, 0, stream>>>(Qb, Kb, Vtb, ctxb);

  // 4) output projection + bias
  gemm_out<<<256, 256, 0, stream>>>(ctxb, Wot, bo, out);
}
